// AttentionHead_68513318306070
// MI455X (gfx1250) — matrix operand order repaired and checked by reading, not compiled, not run on hardware
//
#include <hip/hip_runtime.h>
#include <hip/hip_bf16.h>

// ---- vector types ----
typedef _Float16 h4   __attribute__((ext_vector_type(4)));
typedef _Float16 h8   __attribute__((ext_vector_type(8)));
typedef _Float16 v16h __attribute__((ext_vector_type(16)));
typedef float    v8f  __attribute__((ext_vector_type(8)));
typedef float    f4   __attribute__((ext_vector_type(4)));

#define WMMA_F16(A, B, Cacc) \
  __builtin_amdgcn_wmma_f32_16x16x32_f16(false, (A), false, (B), (short)0, (Cacc), false, false)

// Load a 16-element f16 fragment as two aligned 16B LDS loads.
__device__ __forceinline__ v16h ld16(const _Float16* p0, const _Float16* p1) {
  h8 a = *(const h8*)p0;
  h8 b = *(const h8*)p1;
  return __builtin_shufflevector(a, b, 0,1,2,3,4,5,6,7,8,9,10,11,12,13,14,15);
}

// Problem sizes (fixed by the reference)
constexpr int T = 256, C = 1024, H = 64;
// LDS strides (f16 elements); multiples of 8 => 16B-aligned rows, bank-spread pads.
constexpr int XS = 40;   // x chunk  [256 x 32]
constexpr int WS = 40;   // W chunk  [64  x 32] transposed [n][k]
constexpr int QS = 72;   // q,k      [256 x 64]
constexpr int VS = 264;  // v^T      [64  x 256]
constexpr int PS = 40;   // P scratch per wave [16 x 32]
constexpr int STG = T * XS + H * WS;   // one staging buffer (x + W)

__global__ __launch_bounds__(256)
void attn_head_kernel(const float* __restrict__ x,
                      const float* __restrict__ Wq,
                      const float* __restrict__ Wk,
                      const float* __restrict__ Wv,
                      float* __restrict__ out)
{
  __shared__ __align__(16) _Float16 sQ[T * QS];        // 36 KB
  __shared__ __align__(16) _Float16 sK[T * QS];        // 36 KB
  __shared__ __align__(16) _Float16 sVt[H * VS];       // 33 KB
  __shared__ __align__(16) _Float16 sStage[2 * STG];   // 50 KB (double-buffered x+W)
  _Float16* sP = sStage;  // attention phase reuses staging: 8 waves x [16][PS]

  const int tid  = threadIdx.x;
  const int lane = tid & 31;
  const int wu   = __builtin_amdgcn_readfirstlane(tid >> 5);  // wave id, scalar
  const int l15  = lane & 15;
  const int hi   = lane >> 4;
  const float NEG_INF = -__builtin_inff();

  const float* xb = x + (size_t)blockIdx.x * T * C;

  // staging thread mapping (constant per thread)
  const int cg = tid & 7;            // x: float4 column group
  const int r0 = tid >> 3;           // x: row 0..31 (+32*i)
  const int wn = tid & 63;           // W: column n
  const int kh = (tid >> 6) * 8;     // W: 8 consecutive k rows per thread

  // ======================= Phase 1: q,k,v projections =======================
  const float* Wmat[3] = { Wq, Wk, Wv };
  for (int p = 0; p < 3; ++p) {
    const float* Wp = Wmat[p];
    v8f acc[2][4];
    #pragma unroll
    for (int rt = 0; rt < 2; ++rt)
      #pragma unroll
      for (int nt = 0; nt < 4; ++nt)
        #pragma unroll
        for (int e = 0; e < 8; ++e) acc[rt][nt][e] = 0.0f;

    // ---- prologue: stage chunk 0 into buffer 0
    {
      _Float16* dX = sStage;
      _Float16* dW = sStage + T * XS;
      #pragma unroll
      for (int i = 0; i < 8; ++i) {
        const int r = r0 + 32 * i;
        f4 xv = *(const f4*)(xb + (size_t)r * C + cg * 4);
        h4 hv = { (_Float16)xv[0], (_Float16)xv[1], (_Float16)xv[2], (_Float16)xv[3] };
        *(h4*)(&dX[r * XS + cg * 4]) = hv;
      }
      float wv[8];
      #pragma unroll
      for (int j = 0; j < 8; ++j) wv[j] = Wp[(size_t)(kh + j) * H + wn];
      h8 hw = { (_Float16)wv[0], (_Float16)wv[1], (_Float16)wv[2], (_Float16)wv[3],
                (_Float16)wv[4], (_Float16)wv[5], (_Float16)wv[6], (_Float16)wv[7] };
      *(h8*)(&dW[wn * WS + kh]) = hw;   // single 16B transposed store
    }
    __syncthreads();

    // ---- pipelined main loop: 32 chunks of K=32
    for (int ki = 0; ki < 32; ++ki) {
      const int buf = ki & 1;
      const _Float16* sX = sStage + buf * STG;
      const _Float16* sW = sX + T * XS;
      _Float16* nX = sStage + (buf ^ 1) * STG;
      _Float16* nW = nX + T * XS;

      // issue next chunk's global loads first (overlap with WMMA below)
      f4 xr[8];
      float wr[8];
      if (ki < 31) {
        const int kc = (ki + 1) * 32;
        #pragma unroll
        for (int i = 0; i < 8; ++i)
          xr[i] = *(const f4*)(xb + (size_t)(r0 + 32 * i) * C + kc + cg * 4);
        #pragma unroll
        for (int j = 0; j < 8; ++j)
          wr[j] = Wp[(size_t)(kc + kh + j) * H + wn];
      }

      // compute on current buffer: wave rows [32*wu, 32*wu+31]
      #pragma unroll
      for (int rt = 0; rt < 2; ++rt) {
        const int row = 32 * wu + 16 * rt + l15;
        const _Float16* ap = &sX[row * XS + hi * 8];
        v16h a = ld16(ap, ap + 16);
        #pragma unroll
        for (int nt = 0; nt < 4; ++nt) {
          const _Float16* bp = &sW[(nt * 16 + l15) * WS + hi * 8];
          v16h bf = ld16(bp, bp + 16);
          acc[rt][nt] = WMMA_F16(a, bf, acc[rt][nt]);
        }
      }

      // convert + store next chunk into other buffer
      if (ki < 31) {
        #pragma unroll
        for (int i = 0; i < 8; ++i) {
          h4 hv = { (_Float16)xr[i][0], (_Float16)xr[i][1],
                    (_Float16)xr[i][2], (_Float16)xr[i][3] };
          *(h4*)(&nX[(r0 + 32 * i) * XS + cg * 4]) = hv;
        }
        h8 hw = { (_Float16)wr[0], (_Float16)wr[1], (_Float16)wr[2], (_Float16)wr[3],
                  (_Float16)wr[4], (_Float16)wr[5], (_Float16)wr[6], (_Float16)wr[7] };
        *(h8*)(&nW[wn * WS + kh]) = hw;   // single 16B transposed store
      }
      __syncthreads();
    }

    // ---- write result to LDS f16 (D layout: M = r + 8*hi, N = l15).
    // Single scalar branch on uniform p (no per-store branching).
    if (p < 2) {
      _Float16* dst = (p == 0) ? sQ : sK;
      const float scl = (p == 0) ? (1.0f / 32.0f) : 1.0f;  // fold 1/sqrt(C) into q
      #pragma unroll
      for (int rt = 0; rt < 2; ++rt)
        #pragma unroll
        for (int nt = 0; nt < 4; ++nt)
          #pragma unroll
          for (int r = 0; r < 8; ++r) {
            const int row = 32 * wu + 16 * rt + 8 * hi + r;
            const int col = nt * 16 + l15;
            dst[row * QS + col] = (_Float16)(acc[rt][nt][r] * scl);
          }
    } else {
      #pragma unroll
      for (int rt = 0; rt < 2; ++rt)
        #pragma unroll
        for (int nt = 0; nt < 4; ++nt)
          #pragma unroll
          for (int r = 0; r < 8; ++r) {
            const int row = 32 * wu + 16 * rt + 8 * hi + r;
            const int col = nt * 16 + l15;
            sVt[col * VS + row] = (_Float16)acc[rt][nt][r];  // v transposed
          }
    }
    __syncthreads();
  }

  // ======================= Phase 2: causal flash attention ==================
  // Balanced mapping: wave wu owns row-tiles {wu, 15-wu} (16 rows each).
  // Row-tile j attends key chunks 0..j/2 (32 keys each) -> equal work per wave.
  _Float16* pScr = &sP[wu * 16 * PS];

  for (int half = 0; half < 2; ++half) {
    const int rtile   = half ? (15 - wu) : wu;      // scalar
    const int rowbase = 16 * rtile;
    const int nch     = rtile / 2 + 1;              // scalar trip count

    // q A-fragments for this row-tile (K over H=64 -> 2 steps)
    v16h aq[2];
    #pragma unroll
    for (int kk = 0; kk < 2; ++kk) {
      const _Float16* ap = &sQ[(rowbase + l15) * QS + kk * 32 + hi * 8];
      aq[kk] = ld16(ap, ap + 16);
    }

    v8f o[4];
    float mrow[8], lrow[8];
    #pragma unroll
    for (int ht = 0; ht < 4; ++ht)
      #pragma unroll
      for (int e = 0; e < 8; ++e) o[ht][e] = 0.0f;
    #pragma unroll
    for (int r = 0; r < 8; ++r) { mrow[r] = NEG_INF; lrow[r] = 0.0f; }

    for (int c = 0; c < nch; ++c) {
      const int k0 = 32 * c;

      // ---- S tiles: s[ct] = q . k^T  (contract H = 2 WMMA per tile)
      v8f s[2];
      #pragma unroll
      for (int ct = 0; ct < 2; ++ct)
        #pragma unroll
        for (int e = 0; e < 8; ++e) s[ct][e] = 0.0f;

      #pragma unroll
      for (int ct = 0; ct < 2; ++ct)
        #pragma unroll
        for (int kk = 0; kk < 2; ++kk) {
          const int key = k0 + ct * 16 + l15;       // B col = key, K = h
          const _Float16* bp = &sK[key * QS + kk * 32 + hi * 8];
          v16h bk = ld16(bp, bp + 16);
          s[ct] = WMMA_F16(aq[kk], bk, s[ct]);
        }

      // ---- causal mask on the diagonal (last) chunk — scalar branch
      if (c == nch - 1) {
        #pragma unroll
        for (int ct = 0; ct < 2; ++ct) {
          const int col = k0 + ct * 16 + l15;
          #pragma unroll
          for (int r = 0; r < 8; ++r) {
            const int row = rowbase + 8 * hi + r;
            if (col > row) s[ct][r] = NEG_INF;
          }
        }
      }

      // ---- online softmax
      #pragma unroll
      for (int r = 0; r < 8; ++r) {
        const float sv0 = s[0][r];
        const float sv1 = s[1][r];
        float vmax = fmaxf(sv0, sv1);
        #pragma unroll
        for (int off = 1; off < 16; off <<= 1)
          vmax = fmaxf(vmax, __shfl_xor(vmax, off, 32));  // 16-lane row reduce
        const float mold = mrow[r];
        const float mnew = fmaxf(mold, vmax);
        const float corr = __expf(mold - mnew);
        const float p0 = __expf(sv0 - mnew);
        const float p1 = __expf(sv1 - mnew);
        float ps = p0 + p1;
        #pragma unroll
        for (int off = 1; off < 16; off <<= 1)
          ps += __shfl_xor(ps, off, 32);
        mrow[r] = mnew;
        lrow[r] = lrow[r] * corr + ps;
        #pragma unroll
        for (int ht = 0; ht < 4; ++ht) o[ht][r] *= corr;
        // stash P (D layout) into per-wave LDS scratch
        pScr[(8 * hi + r) * PS + l15]      = (_Float16)p0;
        pScr[(8 * hi + r) * PS + 16 + l15] = (_Float16)p1;
      }

      // ---- reload P as A fragment [16 x 32], accumulate O += P . V
      const _Float16* pp = &pScr[l15 * PS + hi * 8];
      v16h pa = ld16(pp, pp + 16);
      #pragma unroll
      for (int ht = 0; ht < 4; ++ht) {
        const _Float16* vp = &sVt[(ht * 16 + l15) * VS + k0 + hi * 8];
        v16h bv = ld16(vp, vp + 16);
        o[ht] = WMMA_F16(pa, bv, o[ht]);
      }
    }

    // ---- finalize rows of this tile
    float inv[8];
    #pragma unroll
    for (int r = 0; r < 8; ++r) inv[r] = 1.0f / lrow[r];
    #pragma unroll
    for (int ht = 0; ht < 4; ++ht)
      #pragma unroll
      for (int r = 0; r < 8; ++r) {
        const int row = rowbase + 8 * hi + r;
        const int col = ht * 16 + l15;
        out[(size_t)blockIdx.x * T * H + (size_t)row * H + col] = o[ht][r] * inv[r];
      }
  }
}

extern "C" void kernel_launch(void* const* d_in, const int* in_sizes, int n_in,
                              void* d_out, int out_size, void* d_ws, size_t ws_size,
                              hipStream_t stream) {
  (void)n_in; (void)out_size; (void)d_ws; (void)ws_size;
  const float* x  = (const float*)d_in[0];
  const float* Wq = (const float*)d_in[1];
  const float* Wk = (const float*)d_in[2];
  const float* Wv = (const float*)d_in[3];
  float* out = (float*)d_out;
  const int Bn = in_sizes[0] / (T * C);   // 1024
  hipLaunchKernelGGL(attn_head_kernel, dim3(Bn), dim3(256), 0, stream,
                     x, Wq, Wk, Wv, out);
}
